// WaterLevelLSTM_69045894250781
// MI455X (gfx1250) — matrix operand order repaired and checked by reading, not compiled, not run on hardware
//
#include <hip/hip_runtime.h>

typedef _Float16 h16;
typedef __attribute__((ext_vector_type(16))) _Float16 v16h;
typedef __attribute__((ext_vector_type(8)))  _Float16 v8h;
typedef __attribute__((ext_vector_type(8)))  float    v8f;

#define T_STEPS 125
#define BATCH   4096
#define HID     64
#define WAVES   8
#define WROW    136   // padded row stride (halfs) for weights in LDS: 272B, 16B-aligned, bank-staggered
#define HROW    72    // padded row stride (halfs) for h-exchange in LDS: 144B, 16B-aligned

// CDNA5 has hardware V_TANH_F32 (ISA lists TANH among transcendental VALU ops).
#if __has_builtin(__builtin_amdgcn_tanhf)
__device__ __forceinline__ float tanh_hw(float x) { return __builtin_amdgcn_tanhf(x); }
#else
__device__ __forceinline__ float tanh_hw(float x) {
    return 2.0f * __builtin_amdgcn_rcpf(1.0f + __expf(-2.0f * x)) - 1.0f;
}
#endif
// sigmoid(x) = 0.5*tanh(x/2) + 0.5  -> mul + tanh + fma
__device__ __forceinline__ float sigm(float x) { return 0.5f * tanh_hw(0.5f * x) + 0.5f; }

// Scheduling fence: ALU/VALU/SALU/WMMA/TRANS may cross, VMEM/DS may not.
// Bounds LDS-fragment hoisting so VGPR allocation stays <= 256 (no vgpr-msb churn).
__device__ __forceinline__ void sched_fence_mem() {
#if __has_builtin(__builtin_amdgcn_sched_barrier)
    __builtin_amdgcn_sched_barrier(0x40F);
#endif
}

// A-fragment (16x32 f16): lane holds row M = lane%16; K pairs per ISA table:
// lanes 0-15 -> K {kofs+0..7, kofs+16..23}, lanes 16-31 -> K {kofs+8..15, kofs+24..31}
__device__ __forceinline__ v16h afrag(const _Float16* rowbase, int rowstride,
                                      int lo, int hi, int kofs) {
    const _Float16* p = rowbase + lo * rowstride + kofs + (hi ? 8 : 0);
    v8h a = *(const v8h*)p;
    v8h b = *(const v8h*)(p + 16);
    return __builtin_shufflevector(a, b, 0,1,2,3,4,5,6,7,8,9,10,11,12,13,14,15);
}

// B-fragment (32x16 f16): lane holds column N = tile*16 + lane%16; lanes 0-15 hold
// K {kofs+0..7, kofs+16..23}, lanes 16-31 hold K {kofs+8..15, kofs+24..31}.
__device__ __forceinline__ v16h bfrag(const _Float16* sw, int tile,
                                      int lo, int hi, int kofs) {
    const _Float16* p = sw + (tile * 16 + lo) * WROW + kofs + (hi ? 8 : 0);
    v8h a = *(const v8h*)p;
    v8h b = *(const v8h*)(p + 16);
    return __builtin_shufflevector(a, b, 0,1,2,3,4,5,6,7,8,9,10,11,12,13,14,15);
}

template <bool LAYER0>
__global__ void __launch_bounds__(WAVES * 32)
lstm_layer_kernel(const float* __restrict__ x0,        // (B,T,1) f32, layer 0 only
                  const _Float16* __restrict__ hs_in,  // (T,B,H) f16, layers 1..3
                  const float* __restrict__ Wih,       // (4H, K_in) f32
                  const float* __restrict__ Whh,       // (4H, H) f32
                  const float* __restrict__ bih,
                  const float* __restrict__ bhh,
                  _Float16* __restrict__ hs_out)       // (T,B,H) f16
{
    __shared__ __align__(16) _Float16 sW[256 * WROW];        // [x-part | h-part] f16
    __shared__ __align__(16) _Float16 sHX[WAVES][16 * HROW]; // wave-private h D->A transpose

    const int tid  = threadIdx.x;
    const int wave = tid >> 5;
    const int lane = tid & 31;
    const int lo   = lane & 15;
    const int hi   = lane >> 4;
    const int rowbase = (blockIdx.x * WAVES + wave) * 16;

    // Stage weights into LDS as f16; row n = gate-output index.
    {
        const int n = tid; // 256 rows, 256 threads
        if (!LAYER0) {
            #pragma unroll 4
            for (int k = 0; k < HID; ++k) sW[n * WROW + k]       = (h16)Wih[n * HID + k];
            #pragma unroll 4
            for (int k = 0; k < HID; ++k) sW[n * WROW + HID + k] = (h16)Whh[n * HID + k];
        } else {
            #pragma unroll 4
            for (int k = 0; k < HID; ++k) sW[n * WROW + k]       = (h16)Whh[n * HID + k];
        }
    }
    __syncthreads();

    // Per-lane bias (D-layout: lane's column n = tile*16 + lo) and layer-0 input weights.
    float sb[16], wz[16];
    #pragma unroll
    for (int tl = 0; tl < 16; ++tl) {
        const int n = tl * 16 + lo;
        sb[tl] = bih[n] + bhh[n];
        wz[tl] = LAYER0 ? Wih[n] : 0.0f; // W_ih0 is (256,1)
    }

    const int KH = LAYER0 ? 0 : HID; // column offset of h-weights inside sW rows
    _Float16* hx = &sHX[wave][0];

    v8f c[4];
    #pragma unroll
    for (int q = 0; q < 4; ++q)
        #pragma unroll
        for (int r = 0; r < 8; ++r) c[q][r] = 0.0f;

    v16h ha0, ha1; // h_{t-1} A-fragments (zero initial state)
    #pragma unroll
    for (int i = 0; i < 16; ++i) { ha0[i] = (h16)0; ha1[i] = (h16)0; }

    for (int t = 0; t < T_STEPS; ++t) {
        v16h xa0, xa1;
        float xv[8];
        if (LAYER0) {
            #pragma unroll
            for (int r = 0; r < 8; ++r)
                xv[r] = x0[(size_t)(rowbase + r + 8 * hi) * T_STEPS + t];
        } else {
            const _Float16* xb = hs_in + ((size_t)t * BATCH + rowbase) * HID;
            xa0 = afrag(xb, HID, lo, hi, 0);
            xa1 = afrag(xb, HID, lo, hi, 32);
            // prefetch next timestep's x tile (global_prefetch_b8); hs is L2-resident
            const int tn = (t + 1 < T_STEPS) ? t + 1 : t;
            __builtin_prefetch(hs_in + ((size_t)tn * BATCH + rowbase + lo) * HID, 0, 1);
        }

        #pragma unroll
        for (int q = 0; q < 4; ++q) {
            v8f acc[4];
            #pragma unroll
            for (int g = 0; g < 4; ++g) {
                const int tile = g * 4 + q; // gate g, column group q
                v8f a;
                if (LAYER0) {
                    #pragma unroll
                    for (int r = 0; r < 8; ++r) a[r] = sb[tile] + xv[r] * wz[tile];
                } else {
                    #pragma unroll
                    for (int r = 0; r < 8; ++r) a[r] = sb[tile];
                    a = __builtin_amdgcn_wmma_f32_16x16x32_f16(false, xa0, false,
                            bfrag(sW, tile, lo, hi, 0),  (short)0, a, false, false);
                    a = __builtin_amdgcn_wmma_f32_16x16x32_f16(false, xa1, false,
                            bfrag(sW, tile, lo, hi, 32), (short)0, a, false, false);
                }
                a = __builtin_amdgcn_wmma_f32_16x16x32_f16(false, ha0, false,
                        bfrag(sW, tile, lo, hi, KH),      (short)0, a, false, false);
                a = __builtin_amdgcn_wmma_f32_16x16x32_f16(false, ha1, false,
                        bfrag(sW, tile, lo, hi, KH + 32), (short)0, a, false, false);
                acc[g] = a;
            }
            // Elementwise LSTM cell update for column group q (i,f,g,o share D layout).
            #pragma unroll
            for (int r = 0; r < 8; ++r) {
                const float ig = sigm(acc[0][r]);
                const float fg = sigm(acc[1][r]);
                const float gg = tanh_hw(acc[2][r]);
                const float og = sigm(acc[3][r]);
                const float cn = fg * c[q][r] + ig * gg;
                const float hn = og * tanh_hw(cn);
                c[q][r] = cn;
                const h16 hh = (h16)hn;
                const int m = r + 8 * hi;
                hx[m * HROW + q * 16 + lo] = hh;
                hs_out[((size_t)t * BATCH + rowbase + m) * HID + q * 16 + lo] = hh;
            }
            sched_fence_mem(); // bound LDS/VMEM hoisting across q-groups (VGPR pressure)
        }
        // D->A transpose of h via wave-private LDS (in-wave dscnt ordering, no barrier)
        ha0 = afrag(hx, HROW, lo, hi, 0);
        ha1 = afrag(hx, HROW, lo, hi, 32);
    }
}

__global__ void fc_head_kernel(const _Float16* __restrict__ hlast, // (B,H) f16
                               const float* __restrict__ fcw,
                               const float* __restrict__ fcb,
                               float* __restrict__ out)
{
    const int b = blockIdx.x * blockDim.x + threadIdx.x;
    if (b >= BATCH) return;
    const _Float16* h = hlast + (size_t)b * HID;
    float s = 0.0f;
    #pragma unroll
    for (int k = 0; k < HID; ++k) s += (float)h[k] * fcw[k];
    out[b] = s + fcb[0];
}

extern "C" void kernel_launch(void* const* d_in, const int* in_sizes, int n_in,
                              void* d_out, int out_size, void* d_ws, size_t ws_size,
                              hipStream_t stream)
{
    const float* x    = (const float*)d_in[0];
    const float* Wih0 = (const float*)d_in[1];
    const float* Whh0 = (const float*)d_in[2];
    const float* bih0 = (const float*)d_in[3];
    const float* bhh0 = (const float*)d_in[4];
    const float* Wih  = (const float*)d_in[5];  // (3,256,64)
    const float* Whh  = (const float*)d_in[6];  // (3,256,64)
    const float* bih  = (const float*)d_in[7];  // (3,256)
    const float* bhh  = (const float*)d_in[8];  // (3,256)
    const float* fcw  = (const float*)d_in[9];
    const float* fcb  = (const float*)d_in[10];
    float* out = (float*)d_out;

    const size_t hsElems = (size_t)T_STEPS * BATCH * HID;
    _Float16* buf0 = (_Float16*)d_ws;
    _Float16* buf1 = buf0 + hsElems;

    dim3 blk(WAVES * 32);
    dim3 grd(BATCH / (WAVES * 16)); // 32 workgroups, 256 waves = B/16

    lstm_layer_kernel<true><<<grd, blk, 0, stream>>>(x, nullptr, Wih0, Whh0, bih0, bhh0, buf0);

    _Float16* cur = buf0;
    _Float16* nxt = buf1;
    for (int l = 0; l < 3; ++l) {
        lstm_layer_kernel<false><<<grd, blk, 0, stream>>>(
            nullptr, cur,
            Wih + (size_t)l * 256 * HID, Whh + (size_t)l * 256 * HID,
            bih + (size_t)l * 256,       bhh + (size_t)l * 256, nxt);
        _Float16* tmp = cur; cur = nxt; nxt = tmp;
    }
    // `cur` holds last layer's hs; fc on h_T
    fc_head_kernel<<<(BATCH + 255) / 256, 256, 0, stream>>>(
        cur + (size_t)(T_STEPS - 1) * BATCH * HID, fcw, fcb, out);
}
